// HRAN_37598143709631
// MI455X (gfx1250) — matrix operand order repaired and checked by reading, not compiled, not run on hardware
//
#include <hip/hip_runtime.h>
#include <hip/hip_bf16.h>

// ---------------------------------------------------------------------------
// HRAN forward for MI455X (gfx1250, wave32, WMMA).
//
// Dense math: basis-RGCN recast as single GEMMs
//   layer1: [50000 x 1728] @ [1728 x 256]   (agg[N,B*192] ++ x0[N,192])
//   layer2: [50000 x 2304] @ [2304 x 256]   (agg[N,B*256] ++ h1[N,256])
//   pred  : [50000 x 256]  @ [256 x 16(12)]
// via v_wmma_f32_16x16x32_bf16 (f32 accumulate).
//
// Irregular math: per-(dst,rel) counting + mean-weighted scatter. The scatter
// is the HBM bottleneck (agg exceeds the 192MB L2), so agg is kept in bf16 and
// scattered with global_atomic_pk_add_bf16: 2 elems/atomic, half the RMW
// bytes, and the GEMM A operand loads bf16 fragments directly (no cvt VALU).
// Gather sources x0 (38MB) / h1 (51MB) fit in L2.
// ---------------------------------------------------------------------------

namespace {
constexpr int cN     = 50000;
constexpr int cE     = 800000;
constexpr int cIN    = 128;
constexpr int cEMB   = 64;
constexpr int cH     = 256;
constexpr int cR     = 12;
constexpr int cB     = 8;
constexpr int cIN1   = 192;                 // cIN + cEMB
constexpr int cK1    = cB * cIN1 + cIN1;    // 1728
constexpr int cK2    = cB * cH   + cH;      // 2304

// workspace layout (bytes, all offsets 256-aligned); total = 374,787,840 B
constexpr size_t OFF_X0  = 0;            // N*192 f32   (38,400,000)
constexpr size_t OFF_H1  = 38400000;     // N*256 f32   (51,200,000)
constexpr size_t OFF_H2  = 89600000;     // N*256 f32   (51,200,000)
constexpr size_t OFF_CNT = 140800000;    // N*12  i32   ( 2,400,000)
constexpr size_t OFF_BT  = 143200000;    // K2*256 bf16 ( 1,179,648)
constexpr size_t OFF_BP  = 144379648;    // 16*256 bf16 (     8,192)
constexpr size_t OFF_AGG = 144387840;    // N*K2 bf16   (230,400,000)
}

typedef __attribute__((ext_vector_type(16))) __bf16 v16bf;
typedef __attribute__((ext_vector_type(8)))  float  v8f;

union FragAB { unsigned u[8]; uint4 q[2]; v16bf v; };
static_assert(sizeof(FragAB) == 32, "frag size");

static __device__ __forceinline__ unsigned pack2_bf16(float a, float b) {
  union { __bf16 h[2]; unsigned u; } r;
  r.h[0] = (__bf16)a;                       // native f32->bf16 cvt path
  r.h[1] = (__bf16)b;
  return r.u;
}
static __device__ __forceinline__ unsigned short to_bf16(float a) {
  union { __bf16 h; unsigned short u; } r;
  r.h = (__bf16)a;
  return r.u;
}
// CDNA5 packed bf16 atomic add (non-returning, STOREcnt-tracked)
static __device__ __forceinline__ void atomic_pk_add_bf16(unsigned short* addr,
                                                          unsigned val) {
  asm volatile("global_atomic_pk_add_bf16 %0, %1, off"
               :: "v"(addr), "v"(val) : "memory");
}

// ---------------------------- helper kernels -------------------------------

__global__ void k_zero_u32(unsigned* p, long long n) {
  long long i = (long long)blockIdx.x * blockDim.x + threadIdx.x;
  if (i < n) p[i] = 0u;
}

__global__ void k_build_x0(const float* __restrict__ x, float* __restrict__ x0) {
  long long t = (long long)blockIdx.x * blockDim.x + threadIdx.x;
  if (t >= (long long)cN * cIN1) return;
  int n = (int)(t / cIN1), c = (int)(t % cIN1);
  x0[t] = (c < cIN) ? x[(size_t)n * cIN + c] : 0.0f;
}

__global__ void k_add_emb(const int* __restrict__ idx, const float* __restrict__ emb,
                          float* __restrict__ x0, int rows) {
  long long t = (long long)blockIdx.x * blockDim.x + threadIdx.x;
  if (t >= (long long)rows * cEMB) return;
  int g = (int)(t / cEMB), j = (int)(t % cEMB);
  x0[(size_t)idx[g] * cIN1 + cIN + j] += emb[t];   // indices are unique rows
}

__global__ void k_count(const int* __restrict__ ei, const int* __restrict__ et,
                        int* __restrict__ cnt) {
  int e = blockIdx.x * blockDim.x + threadIdx.x;
  if (e >= cE) return;
  atomicAdd(&cnt[ei[cE + e] * cR + et[e]], 1);
}

// one wave per edge (grid-stride): gather x[src] once, scatter 8 basis coeffs
// as packed-bf16 atomics (2 features per op).
__global__ void k_scatter(const int* __restrict__ ei, const int* __restrict__ et,
                          const int* __restrict__ cnt, const float* __restrict__ comp,
                          const float* __restrict__ X,
                          unsigned short* __restrict__ agg, int in, int K) {
  int wid  = (int)(((long long)blockIdx.x * blockDim.x + threadIdx.x) >> 5);
  int lane = threadIdx.x & 31;
  int nW   = (gridDim.x * blockDim.x) >> 5;
  for (int e = wid; e < cE; e += nW) {
    int s = ei[e], d = ei[cE + e], r = et[e];
    float w = 1.0f / fmaxf((float)cnt[d * cR + r], 1.0f);
    float c[cB];
#pragma unroll
    for (int b = 0; b < cB; ++b) c[b] = comp[r * cB + b] * w;
    const float* xs = X + (size_t)s * in;
    unsigned short* ad = agg + (size_t)d * K;
    for (int p = lane; p < (in >> 1); p += 32) {
      float2 xv = *(const float2*)(xs + 2 * p);
#pragma unroll
      for (int b = 0; b < cB; ++b) {
        unsigned val = pack2_bf16(c[b] * xv.x, c[b] * xv.y);
        atomic_pk_add_bf16(ad + b * in + 2 * p, val);
      }
    }
  }
}

__global__ void k_copy_tail(unsigned short* __restrict__ agg,
                            const float* __restrict__ X, int in, int K) {
  long long t = (long long)blockIdx.x * blockDim.x + threadIdx.x;
  if (t >= (long long)cN * in) return;
  int n = (int)(t / in), i = (int)(t % in);
  agg[(size_t)n * K + cB * in + i] = to_bf16(X[t]);
}

// W^T in bf16: BT[n][k] = (k < Bin ? basis[k][n] : root[k-Bin][n])
__global__ void k_build_WT(unsigned short* __restrict__ BT, const float* __restrict__ basis,
                           const float* __restrict__ root, int K, int Bin) {
  long long t = (long long)blockIdx.x * blockDim.x + threadIdx.x;
  if (t >= (long long)cH * K) return;
  int n = (int)(t / K), k = (int)(t % K);
  float v = (k < Bin) ? basis[(size_t)k * cH + n] : root[(size_t)(k - Bin) * cH + n];
  BT[t] = to_bf16(v);
}

__global__ void k_build_WpredT(unsigned short* __restrict__ BP,
                               const float* __restrict__ pw) {
  int t = blockIdx.x * blockDim.x + threadIdx.x;
  if (t >= 16 * cH) return;
  int c = t / cH, k = t % cH;
  BP[t] = (c < 12) ? to_bf16(pw[(size_t)k * 12 + c]) : (unsigned short)0;
}

// --------------------------- WMMA GEMM kernels -----------------------------
// block = 256 thr = 8 waves; block tile = 64 rows x 256 cols.
// wave w: row strip (w&3)*16, col half (w>>2)*128 -> 8 accumulator tiles.
// A frag (16x32 bf16): lane half hi: K run  {kb+hi*16..+15} (row = lane&15)
// B frag (32x16 bf16): lane half hi: K run  {kb+hi*16..+15} (col = lane&15)

template <int K>
__global__ void __launch_bounds__(256) k_gemm_wmma(
    const unsigned short* __restrict__ A, const unsigned short* __restrict__ BT,
    const float* __restrict__ bias, float* __restrict__ Hout, int M) {
  const int wave = threadIdx.x >> 5, lane = threadIdx.x & 31;
  const int rstrip = wave & 3, nhalf = wave >> 2;
  const int rowbase = blockIdx.x * 64 + rstrip * 16;
  const int hi = lane >> 4, l15 = lane & 15;

  int arowi = rowbase + l15;
  if (arowi >= M) arowi = M - 1;                 // clamp (stores are guarded)
  const unsigned short* arow = A + (size_t)arowi * K;

  v8f acc[8];
#pragma unroll
  for (int t = 0; t < 8; ++t) acc[t] = (v8f){0.f, 0.f, 0.f, 0.f, 0.f, 0.f, 0.f, 0.f};

  for (int kb = 0; kb < K; kb += 32) {
    __builtin_prefetch(arow + kb + 32, 0, 0);    // -> global_prefetch_b8
    FragAB af;                                   // bf16 already in frag order
    af.q[0] = *(const uint4*)(arow + kb + hi * 16);
    af.q[1] = *(const uint4*)(arow + kb + hi * 16 + 8);
#pragma unroll
    for (int t = 0; t < 8; ++t) {
      const int col = nhalf * 128 + t * 16 + l15;
      const uint4* bp = (const uint4*)(BT + (size_t)col * K + kb + hi * 16);
      FragAB bfr;
      bfr.q[0] = bp[0];
      bfr.q[1] = bp[1];
      acc[t] = __builtin_amdgcn_wmma_f32_16x16x32_bf16(
          false, af.v, false, bfr.v, (short)0, acc[t], false, false);
    }
  }

#pragma unroll
  for (int t = 0; t < 8; ++t) {
    const int col = nhalf * 128 + t * 16 + l15;
    const float bc = bias[col];
#pragma unroll
    for (int j = 0; j < 8; ++j) {
      int r = rowbase + hi * 8 + j;                 // D: M = hi*8 + j
      if (r < M) Hout[(size_t)r * cH + col] = acc[t][j] + bc;
    }
  }
}

// prediction head: one wave computes 16 rows x 16 cols (12 valid), K = 256
__global__ void __launch_bounds__(256) k_pred_wmma(
    const float* __restrict__ Hm, const unsigned short* __restrict__ BP,
    const float* __restrict__ pb, float* __restrict__ out, int M) {
  const int wave = threadIdx.x >> 5, lane = threadIdx.x & 31;
  const int rowbase = (blockIdx.x * 8 + wave) * 16;
  const int hi = lane >> 4, l15 = lane & 15;

  int arowi = rowbase + l15;
  if (arowi >= M) arowi = M - 1;
  const float* arow = Hm + (size_t)arowi * cH;
  const float pbv = pb[l15 < 12 ? l15 : 0];          // hoisted, uniform-guarded

  v8f acc = (v8f){0.f, 0.f, 0.f, 0.f, 0.f, 0.f, 0.f, 0.f};
  for (int kb = 0; kb < cH; kb += 32) {
    FragAB af;
    {
      const float4* p0 = (const float4*)(arow + kb + hi * 16);
      const float4* p1 = (const float4*)(arow + kb + hi * 16 + 8);
      float4 f0 = p0[0], f1 = p0[1], f2 = p1[0], f3 = p1[1];
      af.u[0] = pack2_bf16(f0.x, f0.y); af.u[1] = pack2_bf16(f0.z, f0.w);
      af.u[2] = pack2_bf16(f1.x, f1.y); af.u[3] = pack2_bf16(f1.z, f1.w);
      af.u[4] = pack2_bf16(f2.x, f2.y); af.u[5] = pack2_bf16(f2.z, f2.w);
      af.u[6] = pack2_bf16(f3.x, f3.y); af.u[7] = pack2_bf16(f3.z, f3.w);
    }
    const uint4* bp = (const uint4*)(BP + (size_t)l15 * cH + kb + hi * 16);
    FragAB bfr;
    bfr.q[0] = bp[0];
    bfr.q[1] = bp[1];
    acc = __builtin_amdgcn_wmma_f32_16x16x32_bf16(
        false, af.v, false, bfr.v, (short)0, acc, false, false);
  }
#pragma unroll
  for (int j = 0; j < 8; ++j) {
    int r = rowbase + hi * 8 + j;
    if (r < M && l15 < 12) out[(size_t)r * 12 + l15] = acc[j] + pbv;
  }
}

// head-attention epilogue: one wave per node, softmax over 4 heads of 64 dims
__global__ void k_attn(float* __restrict__ Hm, const float* __restrict__ att, int M) {
  int gw = (int)(((long long)blockIdx.x * blockDim.x + threadIdx.x) >> 5);
  int lane = threadIdx.x & 31;
  if (gw >= M) return;
  float* hr = Hm + (size_t)gw * cH;
  int head = lane >> 3;
  float4* hp = (float4*)(hr + lane * 8);
  float4 v0 = hp[0], v1 = hp[1];
  const float4* ap = (const float4*)(att + head * 64 + (lane & 7) * 8);
  float4 a0 = ap[0], a1 = ap[1];
  float s = v0.x * a0.x + v0.y * a0.y + v0.z * a0.z + v0.w * a0.w +
            v1.x * a1.x + v1.y * a1.y + v1.z * a1.z + v1.w * a1.w;
  s += __shfl_xor(s, 1); s += __shfl_xor(s, 2); s += __shfl_xor(s, 4);
  float s0 = __shfl(s, 0), s1 = __shfl(s, 8), s2 = __shfl(s, 16), s3 = __shfl(s, 24);
  float mx = fmaxf(fmaxf(s0, s1), fmaxf(s2, s3));
  float e0 = __expf(s0 - mx), e1 = __expf(s1 - mx),
        e2 = __expf(s2 - mx), e3 = __expf(s3 - mx);
  float inv = 1.0f / (e0 + e1 + e2 + e3);
  float al = (head == 0 ? e0 : head == 1 ? e1 : head == 2 ? e2 : e3) * inv;
  v0.x *= al; v0.y *= al; v0.z *= al; v0.w *= al;
  v1.x *= al; v1.y *= al; v1.z *= al; v1.w *= al;
  hp[0] = v0; hp[1] = v1;
}

// ------------------------------- launcher ----------------------------------

extern "C" void kernel_launch(void* const* d_in, const int* in_sizes, int n_in,
                              void* d_out, int out_size, void* d_ws, size_t ws_size,
                              hipStream_t stream) {
  (void)n_in; (void)out_size; (void)ws_size;

  const float* x        = (const float*)d_in[0];
  const int*   ei       = (const int*)d_in[1];
  const int*   et       = (const int*)d_in[2];
  const int*   gene_idx = (const int*)d_in[3];
  const int*   path_idx = (const int*)d_in[4];
  const float* gene_emb = (const float*)d_in[5];
  const float* path_emb = (const float*)d_in[6];
  const float* comp1 = (const float*)d_in[7];
  const float* basis1 = (const float*)d_in[8];
  const float* root1 = (const float*)d_in[9];
  const float* bias1 = (const float*)d_in[10];
  const float* att1 = (const float*)d_in[11];
  const float* comp2 = (const float*)d_in[12];
  const float* basis2 = (const float*)d_in[13];
  const float* root2 = (const float*)d_in[14];
  const float* bias2 = (const float*)d_in[15];
  const float* att2 = (const float*)d_in[16];
  const float* pred_w = (const float*)d_in[17];
  const float* pred_b = (const float*)d_in[18];
  float* out = (float*)d_out;

  char* wsb = (char*)d_ws;
  float* x0  = (float*)(wsb + OFF_X0);
  float* h1  = (float*)(wsb + OFF_H1);
  float* h2  = (float*)(wsb + OFF_H2);
  int*   cnt = (int*)  (wsb + OFF_CNT);
  unsigned short* BT  = (unsigned short*)(wsb + OFF_BT);
  unsigned short* BP  = (unsigned short*)(wsb + OFF_BP);
  unsigned short* agg = (unsigned short*)(wsb + OFF_AGG);

  auto cdiv = [](long long a, long long b) { return (unsigned)((a + b - 1) / b); };
  const int NG = in_sizes ? in_sizes[3] : 20000;
  const int NP = in_sizes ? in_sizes[4] : 2000;

  // node features + embeddings
  k_build_x0<<<cdiv((long long)cN * cIN1, 256), 256, 0, stream>>>(x, x0);
  k_add_emb<<<cdiv((long long)NG * cEMB, 256), 256, 0, stream>>>(gene_idx, gene_emb, x0, NG);
  k_add_emb<<<cdiv((long long)NP * cEMB, 256), 256, 0, stream>>>(path_idx, path_emb, x0, NP);

  // per-(dst, rel) edge counts
  k_zero_u32<<<cdiv((long long)cN * cR, 256), 256, 0, stream>>>((unsigned*)cnt, (long long)cN * cR);
  k_count<<<cdiv(cE, 256), 256, 0, stream>>>(ei, et, cnt);

  // ---- layer 1 ----
  k_zero_u32<<<cdiv((long long)cN * cK1 / 2, 256), 256, 0, stream>>>(
      (unsigned*)agg, (long long)cN * cK1 / 2);
  k_copy_tail<<<cdiv((long long)cN * cIN1, 256), 256, 0, stream>>>(agg, x0, cIN1, cK1);
  k_scatter<<<1024, 256, 0, stream>>>(ei, et, cnt, comp1, x0, agg, cIN1, cK1);
  k_build_WT<<<cdiv((long long)cH * cK1, 256), 256, 0, stream>>>(BT, basis1, root1, cK1, cB * cIN1);
  k_gemm_wmma<cK1><<<cdiv(cN, 64), 256, 0, stream>>>(agg, BT, bias1, h1, cN);
  k_attn<<<cdiv((long long)cN * 32, 256), 256, 0, stream>>>(h1, att1, cN);

  // ---- layer 2 ----
  k_zero_u32<<<cdiv((long long)cN * cK2 / 2, 256), 256, 0, stream>>>(
      (unsigned*)agg, (long long)cN * cK2 / 2);
  k_copy_tail<<<cdiv((long long)cN * cH, 256), 256, 0, stream>>>(agg, h1, cH, cK2);
  k_scatter<<<1024, 256, 0, stream>>>(ei, et, cnt, comp2, h1, agg, cH, cK2);
  k_build_WT<<<cdiv((long long)cH * cK2, 256), 256, 0, stream>>>(BT, basis2, root2, cK2, cB * cH);
  k_gemm_wmma<cK2><<<cdiv(cN, 64), 256, 0, stream>>>(agg, BT, bias2, h2, cN);
  k_attn<<<cdiv((long long)cN * 32, 256), 256, 0, stream>>>(h2, att2, cN);

  // ---- prediction head ----
  k_build_WpredT<<<cdiv(16 * cH, 256), 256, 0, stream>>>(BP, pred_w);
  k_pred_wmma<<<cdiv(cN, 128), 256, 0, stream>>>(h2, BP, pred_b, out, cN);
}
